// GraphAttentionLayer_45853070852230
// MI455X (gfx1250) — matrix operand order repaired and checked by reading, not compiled, not run on hardware
//
#include <hip/hip_runtime.h>

typedef __attribute__((ext_vector_type(16))) _Float16 v16h;
typedef __attribute__((ext_vector_type(8)))  float    v8f;

#define ALPHA  0.2f
#define MASKV  (-9.0e15f)
#define NEGBIG (-1.0e30f)   // exp() underflows to exactly 0
#define NN     4096         // 2N
#define NH     2048         // N
#define NBAT   8

// ---------------------------------------------------------------------------
// Kernel 1: Wh = concat(ht,h) @ W ; Wh1 = Wh@a1 ; Wh2 = Wh@a2 ;
// also store Wh as f16 pre-swizzled into WMMA B-fragment layout:
//   fragment (b, jt, ot): element (k=j_off, n=o%16) -> lane = n + 16*((k%16)/8),
//   half h = (k/16)*8 + k%8.  One wave handles one 32-row j-tile.
// ---------------------------------------------------------------------------
__global__ __launch_bounds__(32) void k1_wh(
    const float* __restrict__ h,  const float* __restrict__ ht,
    const float* __restrict__ W,  const float* __restrict__ a1,
    const float* __restrict__ a2,
    float* __restrict__ wh1, float* __restrict__ wh2,
    _Float16* __restrict__ whfrag)
{
  __shared__ __align__(16) _Float16 lds[32 * 66];   // 32 rows x 64 halves, pad->33 dwords
  const int lane = threadIdx.x;
  const int blk  = blockIdx.x;
  const int b    = blk >> 7;        // / 128
  const int jt   = blk & 127;
  const int j    = jt * 32 + lane;

  const float* src = (j < NH) ? (ht + ((size_t)b * NH + j) * 64)
                              : (h  + ((size_t)b * NH + (j - NH)) * 64);

  float wh[64];
#pragma unroll
  for (int o = 0; o < 64; ++o) wh[o] = 0.f;

  for (int i = 0; i < 64; ++i) {
    float x = src[i];
    const float4* wr = (const float4*)(W + i * 64);   // wave-uniform rows
#pragma unroll
    for (int q = 0; q < 16; ++q) {
      float4 wv = wr[q];
      wh[4*q+0] += x * wv.x;  wh[4*q+1] += x * wv.y;
      wh[4*q+2] += x * wv.z;  wh[4*q+3] += x * wv.w;
    }
  }

  float s1 = 0.f, s2 = 0.f;
#pragma unroll
  for (int o = 0; o < 64; ++o) { s1 += wh[o] * a1[o]; s2 += wh[o] * a2[o]; }
  wh1[b * NN + j] = s1;
  wh2[b * NN + j] = s2;

#pragma unroll
  for (int o = 0; o < 64; ++o) lds[lane * 66 + o] = (_Float16)wh[o];
  __syncthreads();

#pragma unroll
  for (int ot = 0; ot < 4; ++ot) {
    v16h frag;
#pragma unroll
    for (int hh = 0; hh < 16; ++hh) {
      int joff = ((hh >> 3) << 4) + ((lane >> 4) << 3) + (hh & 7); // k = 16*(h/8) + 8*(lane/16) + h%8
      int o    = ot * 16 + (lane & 15);          // n = lane%16
      frag[hh] = lds[joff * 66 + o];             // conflict-free gather
    }
    *((v16h*)whfrag + (((size_t)(b * 128 + jt) * 4 + ot) * 32 + lane)) = frag;
  }
}

// ---------------------------------------------------------------------------
// Kernel 2: per-column softmax stats over masked scores.
//   m_j = max_i a(i,j),  s_j = sum_i exp(a - m).   colp[j] = {Wh2_j, m_j + ln s_j}
// Thread-per-column -> adj reads are coalesced 128B rows across the warp; adj
// lives in L2 (64MB < 192MB) so the two passes stay on-chip.
// ---------------------------------------------------------------------------
__global__ __launch_bounds__(256) void k2_colstats(
    const float* __restrict__ adj,
    const float* __restrict__ wh1, const float* __restrict__ wh2,
    float2* __restrict__ colp)
{
  const int b = blockIdx.y;
  const int j = blockIdx.x * 256 + threadIdx.x;
  const float* w1 = wh1 + b * NN;
  const float  w2 = wh2[b * NN + j];

  float m = -3.0e38f;
  for (int i = 0; i < NN; ++i) {
    float av = adj[(size_t)i * NN + j];
    float e  = w1[i] + w2;
    e = (e > 0.f) ? e : ALPHA * e;
    float a = (av > 0.f) ? e : MASKV;
    m = fmaxf(m, a);
  }
  float s = 0.f;
  for (int i = 0; i < NN; ++i) {
    float av = adj[(size_t)i * NN + j];
    float e  = w1[i] + w2;
    e = (e > 0.f) ? e : ALPHA * e;
    float a = (av > 0.f) ? e : MASKV;
    s += __expf(a - m);            // unconditional exp -> branchless
  }
  colp[b * NN + j] = make_float2(w2, m + __logf(s));
}

// ---------------------------------------------------------------------------
// Kernel 3: hp = P @ Wh fused, ELU + concat epilogue.
// 128 threads (4 waves) per WG own a 64-row i-tile; loop over 128 j-tiles of 32.
// Phase A: compute P tile (f16) into LDS in exact WMMA A-fragment layout
//   A element (m, k): lane = m + 16*((k%16)/8), half h = (k/16)*8 + k%8.
// Phase B: each wave reads its contiguous v16h A-fragment + 4 global v16h
//   B-fragments and issues 4x v_wmma_f32_16x16x32_f16. Double-buffered LDS,
//   one barrier per iteration; fragment loads issued before the exp work so
//   memory latency overlaps the transcendentals.
// ---------------------------------------------------------------------------
__global__ __launch_bounds__(128) void k3_gat(
    const float*  __restrict__ adj,  const float* __restrict__ wh1,
    const float2* __restrict__ colp, const _Float16* __restrict__ whfrag,
    float* __restrict__ out)
{
  // 2 buffers x 4 waves x 32 lanes x 24 halves (48B lane stride)
  __shared__ __align__(16) _Float16 pb[2][3072];

  const int b  = blockIdx.y;
  const int i0 = blockIdx.x * 64;
  const int t  = threadIdx.x;

  // ---- Phase-A identity: 16 consecutive rows x 8 float4-columns ----
  const int f4 = t & 7;          // which float4 of the 32-wide j tile
  const int rl = t >> 3;         // row 0..15 (handles rows rl + 16p, p=0..3)
  const int joffb = 4 * f4;                          // j_off base: 0,4,...,28
  const int g  = (joffb & 15) >> 3;                  // lane-group bit
  const int H0 = ((joffb >> 4) << 3) + (joffb & 7);  // half base in {0,4,8,12}
  const int wbase = rl * 24 + g * 384 + H0;          // + p*768

  float rv[4];
#pragma unroll
  for (int p = 0; p < 4; ++p) rv[p] = wh1[b * NN + i0 + rl + 16 * p];

  // ---- Phase-B identity ----
  const int w    = t >> 5;
  const int lane = t & 31;

  v8f acc[4] = {};

  const float* adjb = adj + (size_t)joffb;
  const v16h*  bfp  = (const v16h*)whfrag;

  // Branchless P-tile producer: select the exp *argument*, exp always executes
  auto writeP = [&](int jtw, int bs) {
    const int j0 = jtw * 32;
    const float4* cp = (const float4*)(colp + (b * NN + j0 + joffb)); // 2 x {wh2, m'}
    float4 c01 = cp[0];
    float4 c23 = cp[1];
    // prefetch adj two tiles ahead (stays resident in WGP$/L2)
    if (jtw + 2 < 128)
      __builtin_prefetch(adjb + (size_t)(i0 + rl) * NN + (jtw + 2) * 32, 0, 1);
#pragma unroll
    for (int p = 0; p < 4; ++p) {
      float4 av = *(const float4*)(adjb + (size_t)(i0 + rl + 16 * p) * NN + j0);
      float r = rv[p];
      float e0 = r + c01.x; e0 = (e0 > 0.f) ? e0 : ALPHA * e0;
      float e1 = r + c01.z; e1 = (e1 > 0.f) ? e1 : ALPHA * e1;
      float e2 = r + c23.x; e2 = (e2 > 0.f) ? e2 : ALPHA * e2;
      float e3 = r + c23.z; e3 = (e3 > 0.f) ? e3 : ALPHA * e3;
      float a0 = (av.x > 0.f) ? (e0 - c01.y) : NEGBIG;
      float a1v = (av.y > 0.f) ? (e1 - c01.w) : NEGBIG;
      float a2v = (av.z > 0.f) ? (e2 - c23.y) : NEGBIG;
      float a3 = (av.w > 0.f) ? (e3 - c23.w) : NEGBIG;
      union { _Float16 hf[4]; uint2 u; } pk;
      pk.hf[0] = (_Float16)__expf(a0);
      pk.hf[1] = (_Float16)__expf(a1v);
      pk.hf[2] = (_Float16)__expf(a2v);
      pk.hf[3] = (_Float16)__expf(a3);
      *(uint2*)&pb[bs][p * 768 + wbase] = pk.u;       // packed b64 LDS store
    }
  };

  writeP(0, 0);

  for (int jt = 0; jt < 128; ++jt) {
    __syncthreads();
    const int cur = jt & 1;

    // Issue all fragment loads first: A from LDS (prev buffer is complete),
    // B from global. They drain while writeP grinds through the exps.
    v16h af = *(const v16h*)&pb[cur][w * 768 + lane * 24];
    const size_t fb = ((size_t)(b * 128 + jt) * 4) * 32 + lane;
    v16h bf0 = bfp[fb + 0 * 32];
    v16h bf1 = bfp[fb + 1 * 32];
    v16h bf2 = bfp[fb + 2 * 32];
    v16h bf3 = bfp[fb + 3 * 32];

    if (jt + 1 < 128) writeP(jt + 1, cur ^ 1);

    acc[0] = __builtin_amdgcn_wmma_f32_16x16x32_f16(false, af, false, bf0,
                                                    (short)0, acc[0], false, false);
    acc[1] = __builtin_amdgcn_wmma_f32_16x16x32_f16(false, af, false, bf1,
                                                    (short)0, acc[1], false, false);
    acc[2] = __builtin_amdgcn_wmma_f32_16x16x32_f16(false, af, false, bf2,
                                                    (short)0, acc[2], false, false);
    acc[3] = __builtin_amdgcn_wmma_f32_16x16x32_f16(false, af, false, bf3,
                                                    (short)0, acc[3], false, false);
  }

  // ---- Epilogue: ELU + concat([hp[:, :N], hp[:, N:]], -1) remap (branchless) ----
#pragma unroll
  for (int ot = 0; ot < 4; ++ot) {
#pragma unroll
    for (int r = 0; r < 8; ++r) {
      float v  = acc[ot][r];
      float ev = __expf(v) - 1.0f;
      v = (v > 0.f) ? v : ev;
      int mm   = ((lane >> 4) << 3) + r;       // D layout: M = 8*(lane/16)+r
      int irow = i0 + w * 16 + mm;
      int o    = ot * 16 + (lane & 15);        // N = lane%16
      int rr   = (irow < NH) ? irow : (irow - NH);
      int oo   = (irow < NH) ? o    : (o + 64);
      out[((size_t)b * NH + rr) * 128 + oo] = v;
    }
  }
}

// ---------------------------------------------------------------------------
extern "C" void kernel_launch(void* const* d_in, const int* in_sizes, int n_in,
                              void* d_out, int out_size, void* d_ws, size_t ws_size,
                              hipStream_t stream) {
  const float* h   = (const float*)d_in[0];
  const float* ht  = (const float*)d_in[1];
  const float* W   = (const float*)d_in[2];
  const float* a1  = (const float*)d_in[3];
  const float* a2  = (const float*)d_in[4];
  const float* adj = (const float*)d_in[5];
  float* out = (float*)d_out;

  char* ws = (char*)d_ws;
  float*    wh1    = (float*)ws;                                   // 128 KB
  float*    wh2    = wh1 + NBAT * NN;                              // 128 KB
  float2*   colp   = (float2*)(wh2 + NBAT * NN);                   // 256 KB
  _Float16* whfrag = (_Float16*)((char*)colp + sizeof(float2) * NBAT * NN); // 4 MB

  k1_wh<<<dim3(NBAT * 128), dim3(32), 0, stream>>>(h, ht, W, a1, a2,
                                                   wh1, wh2, whfrag);
  k2_colstats<<<dim3(NN / 256, NBAT), dim3(256), 0, stream>>>(adj, wh1, wh2, colp);
  k3_gat<<<dim3(NN / 64, NBAT), dim3(128), 0, stream>>>(adj, wh1, colp, whfrag, out);
}
